// Model_43052752175201
// MI455X (gfx1250) — matrix operand order repaired and checked by reading, not compiled, not run on hardware
//
#include <hip/hip_runtime.h>
#include <math.h>
#include <stdint.h>

typedef __attribute__((ext_vector_type(16))) _Float16 v16h;
typedef __attribute__((ext_vector_type(8)))  _Float16 v8h;
typedef __attribute__((ext_vector_type(8)))  float    v8f;

#define B_      4
#define L_      2048
#define BL_     (B_ * L_)
#define DM_     128
#define DI_     256
#define DS_     16
#define DTR_    8
#define DCONV_  4
#define NH_     8
#define HD_     16
#define DOUT_   4
#define EPSF    1e-5f

// ---------------------------------------------------------------- helpers
__device__ __forceinline__ float dev_silu(float v) { return v / (1.f + __expf(-v)); }
__device__ __forceinline__ float dev_softplus(float v) {
    if (v > 20.f) return v;
    if (v < -20.f) return __expf(v);
    return log1pf(__expf(v));
}

// ---------------------------------------------------------------- input proj (K=7, scalar)
__global__ void k_input_proj(const float* __restrict__ x, const float* __restrict__ W,
                             const float* __restrict__ b, float* __restrict__ h) {
    int idx = blockIdx.x * blockDim.x + threadIdx.x;
    if (idx >= BL_ * DM_) return;
    int row = idx >> 7, c = idx & 127;
    const float* xr = x + (size_t)row * 7;
    const float* wr = W + (size_t)c * 7;
    float acc = b[c];
    #pragma unroll
    for (int i = 0; i < 7; ++i) acc += xr[i] * wr[i];
    h[idx] = acc;
}

// ---------------------------------------------------------------- f32 -> f16 convert with zero padding
__global__ void k_cvt_pad(const float* __restrict__ in, _Float16* __restrict__ out,
                          int n, int ntot) {
    int i = blockIdx.x * blockDim.x + threadIdx.x;
    if (i < n) out[i] = (_Float16)in[i];
    else if (i < ntot) out[i] = (_Float16)0.f;
}

// ---------------------------------------------------------------- layernorm (one row per block, D=128)
__global__ __launch_bounds__(128) void k_layernorm(const float* __restrict__ in,
                                                   const float* __restrict__ w,
                                                   const float* __restrict__ b,
                                                   float* __restrict__ out) {
    __shared__ float red[128];
    int row = blockIdx.x, t = threadIdx.x;
    float v = in[(size_t)row * DM_ + t];
    red[t] = v; __syncthreads();
    for (int s = 64; s > 0; s >>= 1) { if (t < s) red[t] += red[t + s]; __syncthreads(); }
    float mu = red[0] * (1.f / DM_); __syncthreads();
    float dv = v - mu;
    red[t] = dv * dv; __syncthreads();
    for (int s = 64; s > 0; s >>= 1) { if (t < s) red[t] += red[t + s]; __syncthreads(); }
    float var = red[0] * (1.f / DM_);
    out[(size_t)row * DM_ + t] = dv * rsqrtf(var + EPSF) * w[t] + b[t];
}

// ---------------------------------------------------------------- rmsnorm (D=128)
__global__ __launch_bounds__(128) void k_rmsnorm(const float* __restrict__ in,
                                                 const float* __restrict__ w,
                                                 float* __restrict__ out) {
    __shared__ float red[128];
    int row = blockIdx.x, t = threadIdx.x;
    float v = in[(size_t)row * DM_ + t];
    red[t] = v * v; __syncthreads();
    for (int s = 64; s > 0; s >>= 1) { if (t < s) red[t] += red[t + s]; __syncthreads(); }
    float ms = red[0] * (1.f / DM_);
    out[(size_t)row * DM_ + t] = v * rsqrtf(ms + EPSF) * w[t];
}

// ---------------------------------------------------------------- WMMA GEMM, fp16 inputs
// out[M,N] = act(A[M,K] @ W[Np,K]^T + bias) + resid   (fp32 accumulate)
// A: f16, lda==K.  W: f16, zero-padded to Np=ceil16(N) rows, ldw==K.
// Block = 4 waves sharing one N-tile; the 16-row W panel is async-staged into LDS
// (global_load_async_to_lds_b128 / ASYNCcnt), B-frags then come from ds_load.
__global__ __launch_bounds__(128)
void k_gemm_f16(const _Float16* __restrict__ A, const _Float16* __restrict__ W,
                const float* __restrict__ bias, const float* __restrict__ resid,
                float* __restrict__ out,
                int M, int N, int K, int ldo, int ldr, int act) {
    __shared__ _Float16 shW[16 * 256];           // 16 rows x K (K<=256)
    const int lane = threadIdx.x;
    const int g    = lane >> 4;                  // lane half
    const int rr   = lane & 15;
    const int ty   = threadIdx.y;
    const int tile_n = blockIdx.x;
    const int tile_m = blockIdx.y * 4 + ty;
    const int tid  = ty * 32 + lane;

    // ---- async stage W panel rows [tile_n*16, +16) x K into LDS ----
    {
        const int cpr   = K >> 3;                // 16B chunks per row
        const int total = cpr << 4;              // 16 rows
        const uint32_t ldsbase = (uint32_t)(uintptr_t)shW;
        for (int c = tid; c < total; c += 128) {
            int row = c / cpr, col = (c - row * cpr) << 3;
            uint64_t ga = (uint64_t)(uintptr_t)(W + (size_t)(tile_n * 16 + row) * K + col);
            uint32_t la = ldsbase + (uint32_t)(row * K + col) * 2u;
            asm volatile("global_load_async_to_lds_b128 %0, %1, off"
                         :: "v"(la), "v"(ga) : "memory");
        }
        asm volatile("s_wait_asynccnt 0x0" ::: "memory");
        __syncthreads();
    }

    const _Float16* ap0 = A + (size_t)(tile_m * 16 + rr) * K;   // A-frag row
    v8f acc = {};
    for (int k0 = 0; k0 < K; k0 += 32) {
        // A-frag: K = 8g+j (j<8), 16+8g+j (j>=8)
        v8h a0 = *(const v8h*)(ap0 + k0 + 8 * g);
        v8h a1 = *(const v8h*)(ap0 + k0 + 16 + 8 * g);
        v16h a = __builtin_shufflevector(a0, a1, 0, 1, 2, 3, 4, 5, 6, 7,
                                                 8, 9, 10, 11, 12, 13, 14, 15);
        // B-frag from LDS: col = rr, K = 8g+j (j<8), 16+8g+j (j>=8)
        const _Float16* sp = shW + rr * K + k0 + 8 * g;
        v8h b0 = *(const v8h*)sp;
        v8h b1 = *(const v8h*)(sp + 16);
        v16h b = __builtin_shufflevector(b0, b1, 0, 1, 2, 3, 4, 5, 6, 7,
                                                 8, 9, 10, 11, 12, 13, 14, 15);
        acc = __builtin_amdgcn_wmma_f32_16x16x32_f16(false, a, false, b, (short)0, acc,
                                                     false, false);
    }
    const int ocol = tile_n * 16 + rr;
    if (ocol >= N) return;
    const float bb = bias ? bias[ocol] : 0.f;
    #pragma unroll
    for (int r = 0; r < 8; ++r) {
        int orow = tile_m * 16 + 8 * g + r;                // C layout: M = 8g + r
        float v = acc[r] + bb;
        if (act == 1) v = dev_silu(v);
        if (resid) v += resid[(size_t)orow * ldr + ocol];
        out[(size_t)orow * ldo + ocol] = v;
    }
}

// ---------------------------------------------------------------- depthwise causal conv (width 4) + SiLU
__global__ void k_conv_silu(const float* __restrict__ xz, const float* __restrict__ cw,
                            const float* __restrict__ cb, float* __restrict__ xact) {
    int idx = blockIdx.x * blockDim.x + threadIdx.x;
    if (idx >= BL_ * DI_) return;
    int d = idx & 255, row = idx >> 8;
    int b = row / L_, l = row % L_;
    float acc = cb[d];
    const float* w = cw + (size_t)d * DCONV_;
    #pragma unroll
    for (int j = 0; j < DCONV_; ++j) {
        int ll = l - (DCONV_ - 1) + j;
        if (ll >= 0) acc += xz[((size_t)b * L_ + ll) * (2 * DI_) + d] * w[j];
    }
    xact[idx] = dev_silu(acc);
}

// ---------------------------------------------------------------- delta = softplus(dt @ dt_W^T + dt_b)
__global__ void k_delta(const float* __restrict__ dbc, const float* __restrict__ dtW,
                        const float* __restrict__ dtb, float* __restrict__ delta) {
    int idx = blockIdx.x * blockDim.x + threadIdx.x;
    if (idx >= BL_ * DI_) return;
    int d = idx & 255, row = idx >> 8;
    const float* dr = dbc + (size_t)row * 40;
    const float* wr = dtW + (size_t)d * DTR_;
    float acc = dtb[d];
    #pragma unroll
    for (int r = 0; r < DTR_; ++r) acc += dr[r] * wr[r];
    delta[idx] = dev_softplus(acc);
}

// ---------------------------------------------------------------- selective scan (register state, LDS-staged B/C)
__global__ __launch_bounds__(256)
void k_scan(const float* __restrict__ dbc, const float* __restrict__ delta,
            const float* __restrict__ xact, const float* __restrict__ A_log,
            const float* __restrict__ Dp, float* __restrict__ y) {
    __shared__ float sh[64 * 32];          // 64 timesteps x (B[16] | C[16])
    const int b = blockIdx.x, d = threadIdx.x;
    float A[DS_], hst[DS_];
    #pragma unroll
    for (int s = 0; s < DS_; ++s) { A[s] = -__expf(A_log[(size_t)d * DS_ + s]); hst[s] = 0.f; }
    const float Dd = Dp[d];
    for (int c = 0; c < L_ / 64; ++c) {
        __syncthreads();
        for (int i = d; i < 64 * 32; i += 256) {
            int step = c * 64 + (i >> 5), j = i & 31;
            sh[i] = dbc[((size_t)b * L_ + step) * 40 + DTR_ + j];
        }
        __syncthreads();
        for (int t = 0; t < 64; ++t) {
            size_t rid = (size_t)b * L_ + c * 64 + t;
            float dl = delta[rid * DI_ + d];
            float xv = xact[rid * DI_ + d];
            const float* bc = &sh[t * 32];
            float acc = 0.f;
            #pragma unroll
            for (int s = 0; s < DS_; ++s) {
                float da = __expf(dl * A[s]);
                hst[s] = da * hst[s] + (dl * bc[s]) * xv;
                acc += hst[s] * bc[16 + s];
            }
            y[rid * DI_ + d] = acc + Dd * xv;
        }
    }
}

// ---------------------------------------------------------------- y *= silu(z)
__global__ void k_gate(float* __restrict__ y, const float* __restrict__ xz) {
    int idx = blockIdx.x * blockDim.x + threadIdx.x;
    if (idx >= BL_ * DI_) return;
    int d = idx & 255, row = idx >> 8;
    y[idx] *= dev_silu(xz[(size_t)row * (2 * DI_) + DI_ + d]);
}

// ---------------------------------------------------------------- h += m
__global__ void k_add(float* __restrict__ h, const float* __restrict__ m) {
    int idx = blockIdx.x * blockDim.x + threadIdx.x;
    if (idx < BL_ * DM_) h[idx] += m[idx];
}

// ---------------------------------------------------------------- flash attention, one wave = 16 query rows
// qkv rows: [q(0..127) | k(128..255) | v(256..383)], head slice h*16.
__global__ __launch_bounds__(128)
void k_attn(const float* __restrict__ qkv, float* __restrict__ yout) {
    __shared__ float ps[4][16 * 33];
    const int lane = threadIdx.x, g = lane >> 4, rr = lane & 15;
    const int wid = threadIdx.y;
    const int bh = blockIdx.y;
    const int b = bh >> 3, h = bh & 7;
    const int qt = blockIdx.x * 4 + wid;
    const size_t base = (size_t)b * L_;
    float* P = ps[wid];
    const v8f zero = {};

    // Q A-frag, zero-padded K>=16, 1/sqrt(hd)=0.25 folded in
    v16h qa;
    {
        const float* qp = qkv + (base + qt * 16 + rr) * 384 + h * HD_;
        #pragma unroll
        for (int j = 0; j < 8; ++j) qa[j] = (_Float16)(g == 0 ? qp[j] * 0.25f : 0.f);   // K=16g+j, hd for g==0
        #pragma unroll
        for (int j = 0; j < 8; ++j) qa[8 + j] = (_Float16)(g == 0 ? qp[8 + j] * 0.25f : 0.f); // K=16g+8+j, hd for g==0
    }

    float mrow[8], lrow[8];
    #pragma unroll
    for (int r = 0; r < 8; ++r) { mrow[r] = -1e30f; lrow[r] = 0.f; }
    v8f o = {};

    for (int kt = 0; kt < L_ / 32; ++kt) {
        v8f s0, s1;
        #pragma unroll
        for (int t = 0; t < 2; ++t) {
            v16h kb;
            if (g == 0) {                        // K rows 0..15 carry hd values, rest zero-pad
                const float* kp = qkv + (base + kt * 32 + t * 16 + rr) * 384 + 128 + h * HD_;
                #pragma unroll
                for (int j = 0; j < 16; ++j) kb[j] = (_Float16)kp[j];
            } else {
                #pragma unroll
                for (int j = 0; j < 16; ++j) kb[j] = (_Float16)0.f;
            }
            v8f s = __builtin_amdgcn_wmma_f32_16x16x32_f16(false, qa, false, kb, (short)0,
                                                           zero, false, false);
            if (t == 0) s0 = s; else s1 = s;
        }
        // online softmax over the 32-key chunk (row stats across 16-lane halves)
        #pragma unroll
        for (int r = 0; r < 8; ++r) {
            float rmax = fmaxf(s0[r], s1[r]);
            #pragma unroll
            for (int msk = 1; msk < 16; msk <<= 1) rmax = fmaxf(rmax, __shfl_xor(rmax, msk, 32));
            float mnew = fmaxf(mrow[r], rmax);
            float p0 = __expf(s0[r] - mnew), p1 = __expf(s1[r] - mnew);
            float rs = p0 + p1;
            #pragma unroll
            for (int msk = 1; msk < 16; msk <<= 1) rs += __shfl_xor(rs, msk, 32);
            float alpha = __expf(mrow[r] - mnew);
            lrow[r] = lrow[r] * alpha + rs;
            mrow[r] = mnew;
            o[r] *= alpha;
            P[(8 * g + r) * 33 + rr]      = p0;   // C-layout -> LDS (row, key)
            P[(8 * g + r) * 33 + 16 + rr] = p1;
        }
        asm volatile("s_wait_dscnt 0" ::: "memory");   // LDS transpose fence (same wave)
        // P A-frag (16 rows x 32 keys)
        v16h pa;
        const float* pr = P + rr * 33;
        #pragma unroll
        for (int j = 0; j < 8; ++j) pa[j]     = (_Float16)pr[16 * g + j];
        #pragma unroll
        for (int j = 0; j < 8; ++j) pa[8 + j] = (_Float16)pr[16 * g + 8 + j];
        // V B-frag (32 keys x 16 hd)
        v16h vb;
        const float* vp = qkv + (base + kt * 32 + g * 16) * 384 + 256 + h * HD_ + rr;
        #pragma unroll
        for (int j = 0; j < 16; ++j) vb[j] = (_Float16)vp[(size_t)j * 384];
        o = __builtin_amdgcn_wmma_f32_16x16x32_f16(false, pa, false, vb, (short)0, o,
                                                   false, false);
        asm volatile("s_wait_dscnt 0" ::: "memory");   // WAR on P across iterations
    }
    #pragma unroll
    for (int r = 0; r < 8; ++r) {
        int row = qt * 16 + 8 * g + r;
        yout[(base + row) * DM_ + h * HD_ + rr] = o[r] / lrow[r];
    }
}

// ---------------------------------------------------------------- decoder (N=4, scalar)
__global__ void k_decode(const float* __restrict__ h, const float* __restrict__ W,
                         const float* __restrict__ b, float* __restrict__ out) {
    int idx = blockIdx.x * blockDim.x + threadIdx.x;
    if (idx >= BL_ * DOUT_) return;
    int o = idx & 3, row = idx >> 2;
    const float* hr = h + (size_t)row * DM_;
    const float* wr = W + (size_t)o * DM_;
    float acc = b[o];
    #pragma unroll 4
    for (int k = 0; k < DM_; ++k) acc += hr[k] * wr[k];
    out[idx] = acc;
}

// ================================================================ host side
struct BlkP {
    const float *ln_m_w, *ln_m_b, *ln_a_w, *ln_a_b, *ln_p_w, *ln_p_b;
    const float *qkv_W, *proj_W, *mlp_W, *mlp_b;
    const float *rms_w, *in_W, *conv_w, *conv_b, *xproj_W, *dt_W, *dt_b, *A_log, *Dv, *out_W;
};

// convert A and W to f16 (W zero-padded to 16-row multiple), then WMMA GEMM
static void gemm(hipStream_t s, const float* A, const float* W, const float* bias,
                 const float* resid, float* out, int M, int N, int K,
                 int ldo, int ldr, int act, _Float16* aF16, _Float16* wF16) {
    const int Np = (N + 15) & ~15;
    int na = M * K, nw = N * K, nwp = Np * K;
    k_cvt_pad<<<(na + 255) / 256, 256, 0, s>>>(A, aF16, na, na);
    k_cvt_pad<<<(nwp + 255) / 256, 256, 0, s>>>(W, wF16, nw, nwp);
    dim3 blk(32, 4);
    dim3 grd(Np / 16, M / 64);
    k_gemm_f16<<<grd, blk, 0, s>>>(aF16, wF16, bias, resid, out, M, N, K, ldo, ldr, act);
}

extern "C" void kernel_launch(void* const* d_in, const int* in_sizes, int n_in,
                              void* d_out, int out_size, void* d_ws, size_t ws_size,
                              hipStream_t stream) {
    (void)in_sizes; (void)n_in; (void)out_size; (void)ws_size;
    // ---- unpack params (depth-first setup_inputs() order) ----
    const float* x       = (const float*)d_in[0];
    // d_in[1] = mask (unused by reference)
    const float* input_W = (const float*)d_in[2];
    const float* input_b = (const float*)d_in[3];
    BlkP blk[2];
    for (int i = 0; i < 2; ++i) {
        int o = 4 + i * 20;
        blk[i].ln_m_w  = (const float*)d_in[o + 0];
        blk[i].ln_m_b  = (const float*)d_in[o + 1];
        blk[i].ln_a_w  = (const float*)d_in[o + 2];
        blk[i].ln_a_b  = (const float*)d_in[o + 3];
        blk[i].ln_p_w  = (const float*)d_in[o + 4];
        blk[i].ln_p_b  = (const float*)d_in[o + 5];
        blk[i].qkv_W   = (const float*)d_in[o + 6];
        blk[i].proj_W  = (const float*)d_in[o + 7];
        blk[i].mlp_W   = (const float*)d_in[o + 8];
        blk[i].mlp_b   = (const float*)d_in[o + 9];
        blk[i].rms_w   = (const float*)d_in[o + 10];
        blk[i].in_W    = (const float*)d_in[o + 11];
        blk[i].conv_w  = (const float*)d_in[o + 12];
        blk[i].conv_b  = (const float*)d_in[o + 13];
        blk[i].xproj_W = (const float*)d_in[o + 14];
        blk[i].dt_W    = (const float*)d_in[o + 15];
        blk[i].dt_b    = (const float*)d_in[o + 16];
        blk[i].A_log   = (const float*)d_in[o + 17];
        blk[i].Dv      = (const float*)d_in[o + 18];
        blk[i].out_W   = (const float*)d_in[o + 19];
    }
    const float* dec_W = (const float*)d_in[4 + 40];
    const float* dec_b = (const float*)d_in[4 + 41];
    float* out = (float*)d_out;

    // ---- workspace layout ----
    float* ws     = (float*)d_ws;
    float* h      = ws;                               // BL*128
    float* m      = h      + (size_t)BL_ * DM_;       // BL*128
    float* buf_ln = m      + (size_t)BL_ * DM_;       // BL*128
    float* xz     = buf_ln + (size_t)BL_ * DM_;       // BL*512 (qkv aliases this)
    float* xact   = xz     + (size_t)BL_ * 2 * DI_;   // BL*256
    float* dbc    = xact   + (size_t)BL_ * DI_;       // BL*48 (ld=40)
    float* delta  = dbc    + (size_t)BL_ * 48;        // BL*256
    float* ybuf   = delta  + (size_t)BL_ * DI_;       // BL*256
    float* attny  = ybuf   + (size_t)BL_ * DI_;       // BL*128
    _Float16* aF16 = (_Float16*)(attny + (size_t)BL_ * DM_);  // BL*256 f16
    _Float16* wF16 = aF16 + (size_t)BL_ * DI_;                // 64K f16
    float* qkv    = xz;                               // alias: xz dead before attention

    const int T256 = 256;
    // 1) input projection
    k_input_proj<<<(BL_ * DM_ + T256 - 1) / T256, T256, 0, stream>>>(x, input_W, input_b, h);

    for (int i = 0; i < 2; ++i) {
        const BlkP& p = blk[i];
        // m = LN_m(h)
        k_layernorm<<<BL_, 128, 0, stream>>>(h, p.ln_m_w, p.ln_m_b, m);
        // ---- mamba: u = rmsnorm(m) ----
        k_rmsnorm<<<BL_, 128, 0, stream>>>(m, p.rms_w, buf_ln);
        // xz = u @ in_W^T                              (8192 x 512 x 128)
        gemm(stream, buf_ln, p.in_W, nullptr, nullptr, xz,
             BL_, 2 * DI_, DM_, 2 * DI_, 0, 0, aF16, wF16);
        // x = silu(conv(x) + b)
        k_conv_silu<<<(BL_ * DI_ + T256 - 1) / T256, T256, 0, stream>>>(xz, p.conv_w, p.conv_b, xact);
        // dbc = x @ xproj_W^T                          (8192 x 40 x 256)
        gemm(stream, xact, p.xproj_W, nullptr, nullptr, dbc,
             BL_, DTR_ + 2 * DS_, DI_, 40, 0, 0, aF16, wF16);
        // delta = softplus(dt @ dt_W^T + dt_b)
        k_delta<<<(BL_ * DI_ + T256 - 1) / T256, T256, 0, stream>>>(dbc, p.dt_W, p.dt_b, delta);
        // selective scan
        k_scan<<<B_, DI_, 0, stream>>>(dbc, delta, xact, p.A_log, p.Dv, ybuf);
        // y *= silu(z)
        k_gate<<<(BL_ * DI_ + T256 - 1) / T256, T256, 0, stream>>>(ybuf, xz);
        // m = y @ out_W^T + m                          (8192 x 128 x 256)
        gemm(stream, ybuf, p.out_W, nullptr, m, m,
             BL_, DM_, DI_, DM_, DM_, 0, aF16, wF16);
        // h = h + m
        k_add<<<(BL_ * DM_ + T256 - 1) / T256, T256, 0, stream>>>(h, m);
        // ---- attention ----
        k_layernorm<<<BL_, 128, 0, stream>>>(h, p.ln_a_w, p.ln_a_b, buf_ln);
        // qkv = ln @ qkv_W^T                           (8192 x 384 x 128)
        gemm(stream, buf_ln, p.qkv_W, nullptr, nullptr, qkv,
             BL_, 3 * DM_, DM_, 3 * DM_, 0, 0, aF16, wF16);
        // flash attention per (b,h), 16 query rows per wave
        k_attn<<<dim3(L_ / 64, B_ * NH_), dim3(32, 4), 0, stream>>>(qkv, attny);
        // h = attny @ proj_W^T + h                     (8192 x 128 x 128)
        gemm(stream, attny, p.proj_W, nullptr, h, h,
             BL_, DM_, DM_, DM_, DM_, 0, aF16, wF16);
        // ---- mlp ----
        k_layernorm<<<BL_, 128, 0, stream>>>(h, p.ln_p_w, p.ln_p_b, buf_ln);
        // h = ln @ mlp_W^T + mlp_b + h                 (8192 x 128 x 128)
        gemm(stream, buf_ln, p.mlp_W, p.mlp_b, h, h,
             BL_, DM_, DM_, DM_, DM_, 0, aF16, wF16);
    }
    // 3) decoder
    k_decode<<<(BL_ * DOUT_ + T256 - 1) / T256, T256, 0, stream>>>(h, dec_W, dec_b, out);
}
